// MPNN_22600117911998
// MI455X (gfx1250) — matrix operand order repaired and checked by reading, not compiled, not run on hardware
//
#include <hip/hip_runtime.h>
#include <hip/hip_bf16.h>
#include <math.h>

typedef __attribute__((ext_vector_type(16))) _Float16 v16h;
typedef __attribute__((ext_vector_type(8)))  _Float16 v8h;
typedef __attribute__((ext_vector_type(8)))  float    v8f;

#define DD    128
#define NTOT  128000
#define E_INT 200000
#define E_TMP 96000

__device__ __forceinline__ v8f v8f_zero() {
  v8f z = {0.f,0.f,0.f,0.f,0.f,0.f,0.f,0.f}; return z;
}

__device__ __forceinline__ v8f wmma16(v16h a, v16h b, v8f c) {
  return __builtin_amdgcn_wmma_f32_16x16x32_f16(false, a, false, b, (short)0, c, false, false);
}

// A-fragment (16x32 f16) loaded from one row pointer (per-lane), row-major.
// Lanes 0-15: M=lane, halves 0-7 = K k0..k0+7, halves 8-15 = K k0+16..k0+23.
// Lanes 16-31: M=lane-16, halves 0-7 = K k0+8.., halves 8-15 = K k0+24..
__device__ __forceinline__ v16h a_frag_row(const _Float16* rowp, int k0, int lane) {
  int koff = (lane >> 4) * 8;
  union { v16h v; v8h h[2]; } u;
  u.h[0] = *(const v8h*)(rowp + k0 + koff);
  u.h[1] = *(const v8h*)(rowp + k0 + 16 + koff);
  return u.v;
}

__device__ __forceinline__ v16h a_frag_rm(const _Float16* base, int ld, int k0, int lane) {
  return a_frag_row(base + (size_t)(lane & 15) * ld, k0, lane);
}

// A-fragment where the logical 256-wide row is concat(r0[0:128], r1[0:128]).
// Each 8-half run lies entirely within one half (k0 multiple of 32).
__device__ __forceinline__ v16h a_frag_concat(const _Float16* r0, const _Float16* r1,
                                              int k0, int lane) {
  int koff = (lane >> 4) * 8;
  int i0 = k0 + koff, i1 = k0 + 16 + koff;
  union { v16h v; v8h h[2]; } u;
  u.h[0] = *(const v8h*)((i0 < DD) ? (r0 + i0) : (r1 + (i0 - DD)));
  u.h[1] = *(const v8h*)((i1 < DD) ? (r0 + i1) : (r1 + (i1 - DD)));
  return u.v;
}

// B-fragment (32x16 f16): N col = n0 + lane%16; halves 0-7 = K k0+8h.., halves 8-15 = K k0+16+8h.. (h = lane/16).
__device__ __forceinline__ v16h b_frag_rm(const _Float16* B, int ldb, int k0, int n0, int lane) {
  int koff = (lane >> 4) * 8;
  const _Float16* p = B + n0 + (lane & 15);
  v16h b;
  #pragma unroll
  for (int s = 0; s < 8; ++s) {
    b[s]     = p[(size_t)(k0 + koff + s) * ldb];
    b[8 + s] = p[(size_t)(k0 + 16 + koff + s) * ldb];
  }
  return b;
}

__device__ __forceinline__ float gelu_f(float x) {
  return 0.5f * x * (1.0f + erff(x * 0.70710678118654752f));
}

// ---------------- small utility kernels ----------------

__global__ void cvt_f32_f16(const float* __restrict__ s, _Float16* __restrict__ d, int n) {
  int i = blockIdx.x * blockDim.x + threadIdx.x;
  if (i < n) d[i] = (_Float16)s[i];
}

__global__ void pad_wout(const float* __restrict__ s, _Float16* __restrict__ d) {
  int i = blockIdx.x * blockDim.x + threadIdx.x;
  if (i < 64 * 16) {
    int r = i >> 4, c = i & 15;
    d[i] = (c < 10) ? (_Float16)s[r * 10 + c] : (_Float16)0.0f;
  }
}

__global__ void count_edges(const int* __restrict__ ni, const int* __restrict__ bi,
                            int E, float* __restrict__ cnt) {
  int e = blockIdx.x * blockDim.x + threadIdx.x;
  if (e < E) {
    atomicAdd(&cnt[bi[e]], 1.0f);
    atomicAdd(&cnt[ni[e]], 1.0f);
  }
}

__global__ void invert_cnt(float* __restrict__ c, int n) {
  int i = blockIdx.x * blockDim.x + threadIdx.x;
  if (i < n) c[i] = 1.0f / fmaxf(c[i], 1.0f);
}

__global__ void embed_k(const int* __restrict__ nin, const float* __restrict__ et,
                        float* __restrict__ h32, _Float16* __restrict__ h16) {
  long long i = (long long)blockIdx.x * blockDim.x + threadIdx.x;   // over NTOT*DD
  long long g = i >> 7;
  int d  = (int)(i & 127);
  int b  = (int)(g / 4000);
  int n  = (int)(g - (long long)b * 4000) % 1000;
  float v = et[nin[b * 1000 + n] * DD + d];
  h32[i] = v;
  h16[i] = (_Float16)v;
}

// ---------------- message MLP + scatter ----------------
// One wave per 16-row tile of the [2E, 256] x-matrix; block = 4 waves.
// Row r<E : x = [h[ni[r]], h[bi[r]]], target bi[r]
// Row r>=E: x = [h[bi[e]], h[ni[e]]], target ni[e]   (e = r - E); target == v in both.

__global__ __launch_bounds__(128) void msg_mlp_k(
    const _Float16* __restrict__ h16,
    const int* __restrict__ ni, const int* __restrict__ bi, int E,
    const _Float16* __restrict__ Win,  const float* __restrict__ bin,
    const _Float16* __restrict__ Whid, const float* __restrict__ bhid,
    const _Float16* __restrict__ Wout, const float* __restrict__ bout,
    float* __restrict__ macc)
{
  __shared__ _Float16 Hs[4][16][72];   // per-wave 16x64 hidden tile, padded stride
  __shared__ int Tg[4][16];

  const int lane = threadIdx.x & 31;
  const int wave = threadIdx.x >> 5;
  const int tile = blockIdx.x * 4 + wave;
  const int r = tile * 16 + (lane & 15);

  int u, v;
  if (r < E) { u = ni[r]; v = bi[r]; }
  else       { int e = r - E; u = bi[e]; v = ni[e]; }
  if (lane < 16) Tg[wave][lane] = v;

  const _Float16* r0 = h16 + (size_t)u * DD;
  const _Float16* r1 = h16 + (size_t)v * DD;

  // Layer 0: X[16x256] @ Win[256x64]
  v8f c[4];
  #pragma unroll
  for (int t = 0; t < 4; ++t) c[t] = v8f_zero();
  #pragma unroll
  for (int k0 = 0; k0 < 2 * DD; k0 += 32) {
    v16h a = a_frag_concat(r0, r1, k0, lane);
    #pragma unroll
    for (int t = 0; t < 4; ++t)
      c[t] = wmma16(a, b_frag_rm(Win, 64, k0, t * 16, lane), c[t]);
  }
  {
    int n0 = lane & 15, mh = (lane >> 4) * 8;
    #pragma unroll
    for (int t = 0; t < 4; ++t) {
      float bv = bin[t * 16 + n0];
      #pragma unroll
      for (int rr = 0; rr < 8; ++rr)
        Hs[wave][mh + rr][t * 16 + n0] = (_Float16)gelu_f(c[t][rr] + bv);
    }
  }
  __syncthreads();

  // 3 hidden layers: H[16x64] @ Whid[64x64]
  for (int l = 0; l < 3; ++l) {
    const _Float16* W = Whid + l * 64 * 64;
    #pragma unroll
    for (int t = 0; t < 4; ++t) c[t] = v8f_zero();
    #pragma unroll
    for (int k0 = 0; k0 < 64; k0 += 32) {
      v16h a = a_frag_rm(&Hs[wave][0][0], 72, k0, lane);
      #pragma unroll
      for (int t = 0; t < 4; ++t)
        c[t] = wmma16(a, b_frag_rm(W, 64, k0, t * 16, lane), c[t]);
    }
    __syncthreads();
    int n0 = lane & 15, mh = (lane >> 4) * 8;
    #pragma unroll
    for (int t = 0; t < 4; ++t) {
      float bv = bhid[l * 64 + t * 16 + n0];
      #pragma unroll
      for (int rr = 0; rr < 8; ++rr)
        Hs[wave][mh + rr][t * 16 + n0] = (_Float16)gelu_f(c[t][rr] + bv);
    }
    __syncthreads();
  }

  // Output layer: H[16x64] @ Wout[64x128], then scatter-add (segment sum)
  v8f o[8];
  #pragma unroll
  for (int t = 0; t < 8; ++t) o[t] = v8f_zero();
  #pragma unroll
  for (int k0 = 0; k0 < 64; k0 += 32) {
    v16h a = a_frag_rm(&Hs[wave][0][0], 72, k0, lane);
    #pragma unroll
    for (int t = 0; t < 8; ++t)
      o[t] = wmma16(a, b_frag_rm(Wout, 128, k0, t * 16, lane), o[t]);
  }
  {
    int n0 = lane & 15, mh = (lane >> 4) * 8;
    #pragma unroll
    for (int rr = 0; rr < 8; ++rr) {
      float* dst = macc + (size_t)Tg[wave][mh + rr] * DD + n0;
      #pragma unroll
      for (int t = 0; t < 8; ++t)
        atomicAdd(dst + t * 16, o[t][rr] + bout[t * 16 + n0]);
    }
  }
}

// ---------------- GRU ----------------
// Block = 4 waves handles one 16-row node tile: m = macc * invc (f16, LDS),
// XA = m @ Wx + bi, HA = h @ Wh + br (24 col-tiles each, split over waves),
// then fused gate math updating h32/h16.

__global__ __launch_bounds__(128) void gru_k(
    float* __restrict__ h32, _Float16* __restrict__ h16,
    const float* __restrict__ macc, const float* __restrict__ invc,
    const _Float16* __restrict__ Wx, const _Float16* __restrict__ Wh,
    const float* __restrict__ bI, const float* __restrict__ bR)
{
  __shared__ _Float16 Ms[16][136];
  __shared__ float XA[16][384];
  __shared__ float HA[16][384];

  const int lane = threadIdx.x & 31;
  const int wave = threadIdx.x >> 5;
  const size_t row0 = (size_t)blockIdx.x * 16;

  for (int i = threadIdx.x; i < 16 * DD; i += 128) {
    int rr = i >> 7, cc = i & 127;
    Ms[rr][cc] = (_Float16)(macc[(row0 + rr) * DD + cc] * invc[row0 + rr]);
  }
  __syncthreads();

  const _Float16* hrow = h16 + row0 * DD;
  #pragma unroll
  for (int j = 0; j < 6; ++j) {
    int n0 = (wave + 4 * j) * 16;    // 24 col-tiles over 4 waves
    v8f ax = v8f_zero(), ah = v8f_zero();
    #pragma unroll
    for (int k0 = 0; k0 < DD; k0 += 32) {
      v16h am = a_frag_rm(&Ms[0][0], 136, k0, lane);
      ax = wmma16(am, b_frag_rm(Wx, 384, k0, n0, lane), ax);
      v16h ahf = a_frag_rm(hrow, DD, k0, lane);
      ah = wmma16(ahf, b_frag_rm(Wh, 384, k0, n0, lane), ah);
    }
    int nn = lane & 15, mh = (lane >> 4) * 8;
    float bx = bI[n0 + nn], bh = bR[n0 + nn];
    #pragma unroll
    for (int rr = 0; rr < 8; ++rr) {
      XA[mh + rr][n0 + nn] = ax[rr] + bx;
      HA[mh + rr][n0 + nn] = ah[rr] + bh;
    }
  }
  __syncthreads();

  for (int i = threadIdx.x; i < 16 * DD; i += 128) {
    int rr = i >> 7, cc = i & 127;
    float z    = 1.0f / (1.0f + expf(-(XA[rr][cc] + HA[rr][cc])));
    float rg   = 1.0f / (1.0f + expf(-(XA[rr][128 + cc] + HA[rr][128 + cc])));
    float cand = tanhf(XA[rr][256 + cc] + rg * HA[rr][256 + cc]);
    size_t gi = (row0 + rr) * DD + cc;
    float hn = z * h32[gi] + (1.0f - z) * cand;
    h32[gi] = hn;
    h16[gi] = (_Float16)hn;
  }
}

// ---------------- readout MLP ----------------
// 32000 rows (b*1000+n), h row = b*4000+n (window 0). One wave per 16-row tile.

__global__ __launch_bounds__(128) void readout_k(
    const _Float16* __restrict__ h16,
    const _Float16* __restrict__ Win,  const float* __restrict__ bin,
    const _Float16* __restrict__ Whid, const float* __restrict__ bhid,
    const _Float16* __restrict__ WoutP, const float* __restrict__ bout,
    float* __restrict__ out)
{
  __shared__ _Float16 Hs[4][16][72];
  const int lane = threadIdx.x & 31;
  const int wave = threadIdx.x >> 5;
  const int tile = blockIdx.x * 4 + wave;
  const int row  = tile * 16 + (lane & 15);
  const int bb   = row / 1000;
  const int n    = row - bb * 1000;
  const _Float16* arow = h16 + ((size_t)bb * 4000 + n) * DD;

  // L0: [16x128] @ Win[128x64]
  v8f c[4];
  #pragma unroll
  for (int t = 0; t < 4; ++t) c[t] = v8f_zero();
  #pragma unroll
  for (int k0 = 0; k0 < DD; k0 += 32) {
    v16h a = a_frag_row(arow, k0, lane);
    #pragma unroll
    for (int t = 0; t < 4; ++t)
      c[t] = wmma16(a, b_frag_rm(Win, 64, k0, t * 16, lane), c[t]);
  }
  {
    int n0 = lane & 15, mh = (lane >> 4) * 8;
    #pragma unroll
    for (int t = 0; t < 4; ++t) {
      float bv = bin[t * 16 + n0];
      #pragma unroll
      for (int rr = 0; rr < 8; ++rr)
        Hs[wave][mh + rr][t * 16 + n0] = (_Float16)gelu_f(c[t][rr] + bv);
    }
  }
  __syncthreads();

  for (int l = 0; l < 3; ++l) {
    const _Float16* W = Whid + l * 64 * 64;
    #pragma unroll
    for (int t = 0; t < 4; ++t) c[t] = v8f_zero();
    #pragma unroll
    for (int k0 = 0; k0 < 64; k0 += 32) {
      v16h a = a_frag_rm(&Hs[wave][0][0], 72, k0, lane);
      #pragma unroll
      for (int t = 0; t < 4; ++t)
        c[t] = wmma16(a, b_frag_rm(W, 64, k0, t * 16, lane), c[t]);
    }
    __syncthreads();
    int n0 = lane & 15, mh = (lane >> 4) * 8;
    #pragma unroll
    for (int t = 0; t < 4; ++t) {
      float bv = bhid[l * 64 + t * 16 + n0];
      #pragma unroll
      for (int rr = 0; rr < 8; ++rr)
        Hs[wave][mh + rr][t * 16 + n0] = (_Float16)gelu_f(c[t][rr] + bv);
    }
    __syncthreads();
  }

  // Output: [16x64] @ WoutP[64x16] (cols 10..15 are zero padding)
  v8f o = v8f_zero();
  #pragma unroll
  for (int k0 = 0; k0 < 64; k0 += 32) {
    v16h a = a_frag_rm(&Hs[wave][0][0], 72, k0, lane);
    o = wmma16(a, b_frag_rm(WoutP, 16, k0, 0, lane), o);
  }
  {
    int n0 = lane & 15, mh = (lane >> 4) * 8;
    if (n0 < 10) {
      float bv = bout[n0];
      #pragma unroll
      for (int rr = 0; rr < 8; ++rr)
        out[(size_t)(tile * 16 + mh + rr) * 10 + n0] = o[rr] + bv;
    }
  }
}

// ---------------- host launcher ----------------

extern "C" void kernel_launch(void* const* d_in, const int* in_sizes, int n_in,
                              void* d_out, int out_size, void* d_ws, size_t ws_size,
                              hipStream_t stream) {
  (void)in_sizes; (void)n_in; (void)out_size; (void)ws_size;

  const int*   node_inputs = (const int*)d_in[0];
  const int*   ni_int = (const int*)d_in[1];
  const int*   bi_int = (const int*)d_in[2];
  const int*   ni_tmp = (const int*)d_in[3];
  const int*   bi_tmp = (const int*)d_in[4];
  const float* etab  = (const float*)d_in[5];
  const float* mWin  = (const float*)d_in[6];
  const float* mbin  = (const float*)d_in[7];
  const float* mWhid = (const float*)d_in[8];
  const float* mbhid = (const float*)d_in[9];
  const float* mWout = (const float*)d_in[10];
  const float* mbout = (const float*)d_in[11];
  const float* rWin  = (const float*)d_in[12];
  const float* rbin  = (const float*)d_in[13];
  const float* rWhid = (const float*)d_in[14];
  const float* rbhid = (const float*)d_in[15];
  const float* rWout = (const float*)d_in[16];
  const float* rbout = (const float*)d_in[17];
  const float* giWx = (const float*)d_in[18];
  const float* giWh = (const float*)d_in[19];
  const float* gibi = (const float*)d_in[20];
  const float* gibr = (const float*)d_in[21];
  const float* gtWx = (const float*)d_in[22];
  const float* gtWh = (const float*)d_in[23];
  const float* gtbi = (const float*)d_in[24];
  const float* gtbr = (const float*)d_in[25];

  char* ws = (char*)d_ws;
  size_t off = 0;
  auto carve = [&](size_t bytes) -> char* {
    off = (off + 255) & ~(size_t)255;
    char* p = ws + off; off += bytes; return p;
  };
  float*    h32     = (float*)carve((size_t)NTOT * DD * 4);
  _Float16* h16     = (_Float16*)carve((size_t)NTOT * DD * 2);
  float*    macc    = (float*)carve((size_t)NTOT * DD * 4);
  float*    inv_int = (float*)carve((size_t)NTOT * 4);
  float*    inv_tmp = (float*)carve((size_t)NTOT * 4);
  _Float16* w_mWin   = (_Float16*)carve(256 * 64 * 2);
  _Float16* w_mWhid  = (_Float16*)carve(3 * 64 * 64 * 2);
  _Float16* w_mWout  = (_Float16*)carve(64 * 128 * 2);
  _Float16* w_rWin   = (_Float16*)carve(128 * 64 * 2);
  _Float16* w_rWhid  = (_Float16*)carve(3 * 64 * 64 * 2);
  _Float16* w_rWoutP = (_Float16*)carve(64 * 16 * 2);
  _Float16* w_giWx   = (_Float16*)carve(128 * 384 * 2);
  _Float16* w_giWh   = (_Float16*)carve(128 * 384 * 2);
  _Float16* w_gtWx   = (_Float16*)carve(128 * 384 * 2);
  _Float16* w_gtWh   = (_Float16*)carve(128 * 384 * 2);

  auto cvt = [&](const float* s, _Float16* d, int n) {
    cvt_f32_f16<<<(n + 255) / 256, 256, 0, stream>>>(s, d, n);
  };
  cvt(mWin,  w_mWin,  256 * 64);
  cvt(mWhid, w_mWhid, 3 * 64 * 64);
  cvt(mWout, w_mWout, 64 * 128);
  cvt(rWin,  w_rWin,  128 * 64);
  cvt(rWhid, w_rWhid, 3 * 64 * 64);
  cvt(giWx, w_giWx, 128 * 384);
  cvt(giWh, w_giWh, 128 * 384);
  cvt(gtWx, w_gtWx, 128 * 384);
  cvt(gtWh, w_gtWh, 128 * 384);
  pad_wout<<<4, 256, 0, stream>>>(rWout, w_rWoutP);

  // segment mean denominators (constant across rounds)
  hipMemsetAsync(inv_int, 0, (size_t)NTOT * 4, stream);
  hipMemsetAsync(inv_tmp, 0, (size_t)NTOT * 4, stream);
  count_edges<<<(E_INT + 255) / 256, 256, 0, stream>>>(ni_int, bi_int, E_INT, inv_int);
  count_edges<<<(E_TMP + 255) / 256, 256, 0, stream>>>(ni_tmp, bi_tmp, E_TMP, inv_tmp);
  invert_cnt<<<(NTOT + 255) / 256, 256, 0, stream>>>(inv_int, NTOT);
  invert_cnt<<<(NTOT + 255) / 256, 256, 0, stream>>>(inv_tmp, NTOT);

  // h = embed_table[node_inputs] repeated over window
  embed_k<<<(NTOT * DD) / 256, 256, 0, stream>>>(node_inputs, etab, h32, h16);

  // T=2 outer rounds of (interaction, temporal)
  for (int step = 0; step < 4; ++step) {
    const bool temporal = (step & 1);
    const int* ni = temporal ? ni_tmp : ni_int;
    const int* bi = temporal ? bi_tmp : bi_int;
    const int  E  = temporal ? E_TMP : E_INT;
    const float* inv = temporal ? inv_tmp : inv_int;
    const _Float16* Wx = temporal ? w_gtWx : w_giWx;
    const _Float16* Wh = temporal ? w_gtWh : w_giWh;
    const float* bI = temporal ? gtbi : gibi;
    const float* bR = temporal ? gtbr : gibr;

    hipMemsetAsync(macc, 0, (size_t)NTOT * DD * 4, stream);
    msg_mlp_k<<<(2 * E) / 64, 128, 0, stream>>>(h16, ni, bi, E,
                                                w_mWin, mbin, w_mWhid, mbhid,
                                                w_mWout, mbout, macc);
    gru_k<<<NTOT / 16, 128, 0, stream>>>(h32, h16, macc, inv, Wx, Wh, bI, bR);
  }

  readout_k<<<32000 / 64, 128, 0, stream>>>(h16, w_rWin, rbin, w_rWhid, rbhid,
                                            w_rWoutP, rbout, (float*)d_out);
}
